// Contrast_single_61890478736011
// MI455X (gfx1250) — compile-verified, matrix operand order checked by reading, not run on hardware
//
#include <hip/hip_runtime.h>
#include <hip/hip_bf16.h>
#include <cstdint>
#include <cstddef>

typedef __attribute__((ext_vector_type(16))) _Float16 v16h;
typedef __attribute__((ext_vector_type(8)))  float    v8f;
typedef __attribute__((ext_vector_type(4)))  _Float16 v4h;
typedef __attribute__((ext_vector_type(4)))  uint32_t v4u;
typedef __attribute__((ext_vector_type(8)))  uint32_t v8u;

#define NROWS 8192
#define DDIM  512
#define BM 128
#define BN 128
#define KH  256            // K-half in elements
#define LDT  520           // A panel LDS row stride (halves): 1024B + 16B TDM pad
#define LDTH 264           // B half-panel LDS row stride (halves): 512B + 16B TDM pad
#define TAU_INV 1.25f      // 1/0.8

// ---------------------------------------------------------------------------
// TDM: DMA a (128 x tile_elems)-half tile of zh into LDS with per-row padding.
// D# per CDNA5 ISA (2D: groups 0+1 only).
//   pad_code: (1<<pad_code) 8-byte units between pads; pad_amount=3 -> 4 DWORDs.
// Tensor dims set equal to the tile (global_addr is the tile start) so no OOB.
// ---------------------------------------------------------------------------
__device__ __forceinline__ void tdm_load_panel(const _Float16* gsrc, uint32_t lds_off,
                                               uint32_t tile_elems, uint32_t pad_code) {
    const uint64_t ga = (uint64_t)(uintptr_t)gsrc;
    v4u g0;
    g0[0] = 1u;                                    // count=1, is_restore=0, no gather
    g0[1] = lds_off;                               // lds_addr (bytes)
    g0[2] = (uint32_t)ga;                          // global_addr[31:0]
    g0[3] = (uint32_t)(ga >> 32) | (2u << 30);     // global_addr[56:32] | type=2
    v8u g1;
    g1[0] = (1u << 16)                             // data_size = 1 (2 bytes)
          | (1u << 20)                             // pad_enable
          | (pad_code << 22)                       // pad_interval
          | (3u << 25);                            // pad_amount: 4 DWORDs (16B)
    g1[1] = tile_elems << 16;                      // tensor_dim0 (bits 79:48)
    g1[2] = (uint32_t)BM << 16;                    // tensor_dim1 = 128 (bits 111:80)
    g1[3] = tile_elems << 16;                      // tile_dim0 (bits 127:112)
    g1[4] = (uint32_t)BM;                          // tile_dim1 = 128, tile_dim2 = 0
    g1[5] = (uint32_t)DDIM;                        // tensor_dim0_stride = 512 (row step)
    g1[6] = 0u;
    g1[7] = 0u;
    asm volatile("tensor_load_to_lds %0, %1" :: "s"(g0), "s"(g1) : "memory");
}

// ---------------------------------------------------------------------------
// Kernel 0: zero the atomically-accumulated buffers (deterministic per call)
// ---------------------------------------------------------------------------
__global__ void init_kernel(float* __restrict__ out,
                            float* __restrict__ tsum,
                            float* __restrict__ psum) {
    int i = blockIdx.x * 256 + threadIdx.x;
    if (i == 0) out[0] = 0.f;
    if (i < NROWS) { tsum[i] = 0.f; psum[i] = 0.f; }
}

// ---------------------------------------------------------------------------
// Kernel 1: per-row L2 normalize z (f32) -> zh (f16).  cos = zh @ zh^T.
// ---------------------------------------------------------------------------
__global__ void __launch_bounds__(128)
rownorm_kernel(const float* __restrict__ z, _Float16* __restrict__ zh) {
    const int row = blockIdx.x;
    const int t   = threadIdx.x;                         // 0..127
    const float4 v = reinterpret_cast<const float4*>(z + (size_t)row * DDIM)[t];
    float ss = v.x * v.x + v.y * v.y + v.z * v.z + v.w * v.w;
#pragma unroll
    for (int off = 16; off; off >>= 1) ss += __shfl_xor(ss, off, 32);
    __shared__ float partial[4];
    if ((t & 31) == 0) partial[t >> 5] = ss;
    __syncthreads();
    const float rinv = rsqrtf(partial[0] + partial[1] + partial[2] + partial[3]);
    v4h h;
    h.x = (_Float16)(v.x * rinv);
    h.y = (_Float16)(v.y * rinv);
    h.z = (_Float16)(v.z * rinv);
    h.w = (_Float16)(v.w * rinv);
    reinterpret_cast<v4h*>(zh + (size_t)row * DDIM)[t] = h;
}

// ---------------------------------------------------------------------------
// Kernel 2: fused WMMA GEMM (zh @ zh^T) + exp(s/tau) + pos mask + row sums.
// Block = 256 threads = 8 waves (2 M-strips x 4 N-strips), tile 128x128xK512.
// B panels split into K-halves, double-buffered in LDS; TDM DMA overlapped
// with WMMA compute via in-order TENSORcnt (s_wait_tensorcnt 1).
// ---------------------------------------------------------------------------
__global__ void __launch_bounds__(256, 1)
nce_gemm_kernel(const _Float16* __restrict__ zh, const int* __restrict__ pos,
                float* __restrict__ tsum, float* __restrict__ psum) {
    extern __shared__ char smem[];
    _Float16* As  = (_Float16*)smem;                // 128 x 520 halves (full K)
    _Float16* Bh0 = As  + BM * LDT;                 // 128 x 264 halves (K-half buf 0)
    _Float16* Bh1 = Bh0 + BM * LDTH;                // 128 x 264 halves (K-half buf 1)
    float*    t_s = (float*)(Bh1 + BM * LDTH);      // 128 row exp-sums
    float*    p_s = t_s + BM;                       // 128 row masked exp-sums

    const int tid     = threadIdx.x;
    const int rowbase = blockIdx.x * BM;
    const uint32_t lds_base = (uint32_t)(uintptr_t)(void*)smem;
    const uint32_t As_off  = lds_base;
    const uint32_t Bh0_off = lds_base + (uint32_t)(BM * LDT  * sizeof(_Float16));
    const uint32_t Bh1_off = Bh0_off  + (uint32_t)(BM * LDTH * sizeof(_Float16));

    for (int i = tid; i < BM; i += 256) { t_s[i] = 0.f; p_s[i] = 0.f; }

    const int lane = tid & 31;
    const int wave = tid >> 5;
    const int wm   = wave & 1;                      // M strip: 0..1 (64 rows)
    const int wn   = wave >> 1;                     // N strip: 0..3 (32 cols)
    const int lrow = lane & 15;                     // row/col within 16-tile
    const int kh   = (lane >> 4) * 8;               // per-lane K sub-offset
    const int moff = (lane >> 4) * 8;               // C-layout M offset

    const _Float16* Afrag  = As  + (wm * 64 + lrow) * LDT  + kh;
    const _Float16* Bfrag0 = Bh0 + (wn * 32 + lrow) * LDTH + kh;
    const _Float16* Bfrag1 = Bh1 + (wn * 32 + lrow) * LDTH + kh;

    const int cb0 = blockIdx.y * 8;                 // first column block of group

    // ---- pipeline prologue: A panel, then B[0] halves; keep h1 in flight ----
    if (tid < 32) {
        tdm_load_panel(zh + (size_t)rowbase * DDIM, As_off, DDIM, 7u);
        tdm_load_panel(zh + (size_t)(cb0 * BN) * DDIM,       Bh0_off, KH, 6u);
        tdm_load_panel(zh + (size_t)(cb0 * BN) * DDIM + KH,  Bh1_off, KH, 6u);
        __builtin_amdgcn_s_wait_tensorcnt(1);       // A + B0.h0 done; B0.h1 in flight
    }
    __syncthreads();

    for (int c = 0; c < 8; ++c) {
        const int colbase = (cb0 + c) * BN;

        v8f acc[4][2] = {};
        v16h a[2][4], b[2][2];

        // fragment loader: half h (A K-offset h*256), local k-step kkl in [0,8)
        auto load_frags = [&](const _Float16* Bfrag, int h, int kkl, int s) {
#pragma unroll
            for (int mt = 0; mt < 4; ++mt) {
                const _Float16* p = Afrag + mt * (16 * LDT) + h * KH + kkl * 32;
                uint4* d = (uint4*)&a[s][mt];
                d[0] = *(const uint4*)p;            // K = kh .. kh+7
                d[1] = *(const uint4*)(p + 16);     // K = kh+16 .. kh+23
            }
#pragma unroll
            for (int nt = 0; nt < 2; ++nt) {
                const _Float16* p = Bfrag + nt * (16 * LDTH) + kkl * 32;
                uint4* d = (uint4*)&b[s][nt];
                d[0] = *(const uint4*)p;
                d[1] = *(const uint4*)(p + 16);
            }
        };
        auto compute_half = [&](const _Float16* Bfrag, int h) {
            load_frags(Bfrag, h, 0, 0);
#pragma unroll
            for (int kkl = 0; kkl < 8; ++kkl) {
                const int cur = kkl & 1;
                if (kkl < 7) load_frags(Bfrag, h, kkl + 1, cur ^ 1);
#pragma unroll
                for (int mt = 0; mt < 4; ++mt)
#pragma unroll
                    for (int nt = 0; nt < 2; ++nt)
                        acc[mt][nt] = __builtin_amdgcn_wmma_f32_16x16x32_f16(
                            false, a[cur][mt], false, b[cur][nt], (short)0,
                            acc[mt][nt], false, false);
            }
        };

        // ---- K-half 0 on Bh0; B[c].h1 DMA is concurrently in flight ----
        compute_half(Bfrag0, 0);
        __syncthreads();                            // all waves done reading Bh0
        if (tid < 32) {
            if (c < 7)                              // stream B[c+1].h0 into Bh0
                tdm_load_panel(zh + (size_t)((cb0 + c + 1) * BN) * DDIM, Bh0_off, KH, 6u);
            __builtin_amdgcn_s_wait_tensorcnt(1);   // B[c].h1 done; next h0 in flight
        }
        __syncthreads();

        // ---- K-half 1 on Bh1; B[c+1].h0 DMA concurrently in flight ----
        compute_half(Bfrag1, 1);

        // ---- epilogue: exp(s/tau), pos mask, 16-lane row reduce, LDS acc ----
#pragma unroll
        for (int mt = 0; mt < 4; ++mt) {
#pragma unroll
            for (int nt = 0; nt < 2; ++nt) {
                const int col = colbase + wn * 32 + nt * 16 + lrow;  // N = lane%16
#pragma unroll
                for (int r = 0; r < 8; ++r) {
                    const int rloc = wm * 64 + mt * 16 + moff + r;   // local row
                    float e = __expf(acc[mt][nt][r] * TAU_INV);
                    const int pv = pos[(size_t)(rowbase + rloc) * NROWS + col];
                    float pe = pv ? e : 0.f;
#pragma unroll
                    for (int off = 8; off; off >>= 1) {              // within 16
                        e  += __shfl_xor(e,  off, 32);
                        pe += __shfl_xor(pe, off, 32);
                    }
                    if (lrow == 0) {
                        atomicAdd(&t_s[rloc], e);
                        atomicAdd(&p_s[rloc], pe);
                    }
                }
            }
        }

        __syncthreads();                            // all waves done reading Bh1
        if (tid < 32 && c < 7) {
            // stream B[c+1].h1 into Bh1; wait for B[c+1].h0 (needed next half0)
            tdm_load_panel(zh + (size_t)((cb0 + c + 1) * BN) * DDIM + KH, Bh1_off, KH, 6u);
            __builtin_amdgcn_s_wait_tensorcnt(1);
        }
        __syncthreads();
    }

    for (int i = tid; i < BM; i += 256) {
        atomicAdd(&tsum[rowbase + i], t_s[i]);
        atomicAdd(&psum[rowbase + i], p_s[i]);
    }
}

// ---------------------------------------------------------------------------
// Kernel 3: loss = mean_i [ log(t_i + 1e-8) - log(p_i) ]
// ---------------------------------------------------------------------------
__global__ void __launch_bounds__(256)
finalize_kernel(const float* __restrict__ tsum, const float* __restrict__ psum,
                float* __restrict__ out) {
    __shared__ float red[256];
    const int i = blockIdx.x * 256 + threadIdx.x;   // 32 blocks x 256 = 8192
    float v = 0.f;
    if (i < NROWS)
        v = (logf(tsum[i] + 1e-8f) - logf(psum[i])) * (1.0f / (float)NROWS);
    red[threadIdx.x] = v;
    __syncthreads();
#pragma unroll
    for (int s = 128; s; s >>= 1) {
        if (threadIdx.x < s) red[threadIdx.x] += red[threadIdx.x + s];
        __syncthreads();
    }
    if (threadIdx.x == 0) atomicAdd(out, red[0]);
}

// ---------------------------------------------------------------------------
extern "C" void kernel_launch(void* const* d_in, const int* in_sizes, int n_in,
                              void* d_out, int out_size, void* d_ws, size_t ws_size,
                              hipStream_t stream) {
    const float* z   = (const float*)d_in[0];       // [8192, 512] f32
    const int*   pos = (const int*)d_in[1];         // [8192, 8192] i32
    float*       out = (float*)d_out;               // scalar

    // workspace: zh (8 MB f16) | tsum (32 KB) | psum (32 KB)
    _Float16* zh   = (_Float16*)d_ws;
    float*    tsum = (float*)((char*)d_ws + (size_t)NROWS * DDIM * sizeof(_Float16));
    float*    psum = tsum + NROWS;

    const size_t smem = (size_t)BM * LDT * sizeof(_Float16)
                      + (size_t)2 * BM * LDTH * sizeof(_Float16)
                      + (size_t)2 * BM * sizeof(float);   // ~264 KB of 320 KB/WGP
    (void)hipFuncSetAttribute((const void*)nce_gemm_kernel,
                              hipFuncAttributeMaxDynamicSharedMemorySize,
                              (int)smem);

    init_kernel<<<(NROWS + 256) / 256 + 1, 256, 0, stream>>>(out, tsum, psum);
    rownorm_kernel<<<NROWS, 128, 0, stream>>>(z, zh);
    dim3 grid(NROWS / BM, 8);                       // 64 row-blocks x 8 col-groups
    nce_gemm_kernel<<<grid, 256, smem, stream>>>(zh, pos, tsum, psum);
    finalize_kernel<<<NROWS / 256, 256, 0, stream>>>(tsum, psum, out);
}
